// CausalSelfAttention_51256139710883
// MI455X (gfx1250) — matrix operand order repaired and checked by reading, not compiled, not run on hardware
//
#include <hip/hip_runtime.h>

typedef unsigned short u16;
typedef unsigned int u32;
typedef __attribute__((ext_vector_type(16))) __bf16 v16bf;
typedef __attribute__((ext_vector_type(8)))  float  v8f;
typedef __attribute__((ext_vector_type(4)))  u32    u32x4;
typedef __attribute__((ext_vector_type(8)))  int    i32x8;
typedef __attribute__((ext_vector_type(4)))  int    i32x4;

#define WMMA_BF16(A, B, C) \
    __builtin_amdgcn_wmma_f32_16x16x32_bf16(false, (A), false, (B), (short)0, (C), false, false)

// ---------------- helpers ----------------

__device__ __forceinline__ u16 f2bf(float f) {
    u32 u = __float_as_uint(f);
    u += 0x7FFFu + ((u >> 16) & 1u);   // round-to-nearest-even
    return (u16)(u >> 16);
}

union FragU { uint4 q[2]; v16bf v; };

// A fragment (16x32 bf16): lane m = lane&15, half = lane>>4.
// elems 0..7  <- k = half*8 + 0..7 ; elems 8..15 <- k = 16 + half*8 + 0..7
__device__ __forceinline__ v16bf load_fragA(const u16* p, int stride, int lane) {
    const int m = lane & 15, h = lane >> 4;
    const u16* r = p + m * stride;
    FragU f;
    f.q[0] = *(const uint4*)(r + h * 8);
    f.q[1] = *(const uint4*)(r + 16 + h * 8);
    return f.v;
}

// B fragment (32x16 bf16): lane n = lane&15, half = lane>>4.
// elems 0..7  <- k = half*8 + 0..7 ; elems 8..15 <- k = 16 + half*8 + 0..7
__device__ __forceinline__ v16bf load_fragB(const u16* p, int stride, int lane) {
    const int n = lane & 15, h = lane >> 4;
    const u16* r = p + n * stride;
    FragU f;
    f.q[0] = *(const uint4*)(r + h * 8);
    f.q[1] = *(const uint4*)(r + 16 + h * 8);
    return f.v;
}

// ---------------- TDM tile load (2-D, bf16, with LDS row padding) ----------------
// Loads a tile of tile_dim0=32 elems (64B) x tile_dim1=128 rows into LDS,
// inserting 16B of pad after every 64B row -> LDS row stride 80B (40 u16).
// D# per CDNA5 ISA.

__device__ __forceinline__ void tdm_load_tile_2d(u32 ldsAddr, const u16* gsrc,
                                                 u32 rowLen /*elements*/, u32 nRows) {
    unsigned long long ga = (unsigned long long)(size_t)gsrc;
    u32x4 g0;
    g0.x = 1u;                                  // count=1 (valid user descriptor)
    g0.y = ldsAddr;                             // LDS byte address
    g0.z = (u32)ga;                             // global_addr[31:0]
    g0.w = (u32)(ga >> 32) | (2u << 30);        // global_addr[56:32] | type=2
    i32x8 g1;
    g1[0] = (int)((1u << 16)    // data_size = 2 bytes
                | (1u << 20)    // pad_enable
                | (3u << 22)    // pad_interval: 16 DWORDs (one 64B row)
                | (3u << 25));  // pad_amount:   4 DWORDs (16B)
    g1[1] = (int)((rowLen & 0xFFFFu) << 16);                        // tensor_dim0[15:0]
    g1[2] = (int)(((rowLen >> 16) & 0xFFFFu) | ((nRows & 0xFFFFu) << 16)); // dim0 hi | dim1 lo
    g1[3] = (int)(((nRows >> 16) & 0xFFFFu) | (32u << 16));         // dim1 hi | tile_dim0=32
    g1[4] = (int)128;                                               // tile_dim1=128, tile_dim2=0
    g1[5] = (int)rowLen;                                            // tensor_dim0_stride lo
    g1[6] = 0;                                                      // stride hi | dim1_stride lo
    g1[7] = 0;
    i32x4 z4 = {0, 0, 0, 0};
#if __clang_major__ >= 23
    i32x8 z8 = {0, 0, 0, 0, 0, 0, 0, 0};
    __builtin_amdgcn_tensor_load_to_lds(g0, g1, z4, z4, z8, 0);
#else
    __builtin_amdgcn_tensor_load_to_lds(g0, g1, z4, z4, 0);
#endif
}

// ---------------- stage 1: fp32 -> bf16 ----------------

__global__ __launch_bounds__(256) void cvt_bf16_kernel(const float* __restrict__ src,
                                                       u16* __restrict__ dst, int n4) {
    int i = blockIdx.x * 256 + threadIdx.x;
    if (i < n4) {
        float4 f = ((const float4*)src)[i];
        uint2 p;
        p.x = (u32)f2bf(f.x) | ((u32)f2bf(f.y) << 16);
        p.y = (u32)f2bf(f.z) | ((u32)f2bf(f.w) << 16);
        ((uint2*)dst)[i] = p;
    }
}

// ---------------- shared GEMM main loop (TDM double-buffered) ----------------
// C[128x128] = A[128x1024] * B[128x1024]^T  (row-major bf16, ld = 1024)
// 8 waves: wm = wave&3 (32 rows each), wn = wave>>2 (64 cols each).
// Wave 0 DMAs the A tile, wave 1 the B tile; buffers ping-pong so the TDM for
// tile kb+1 overlaps WMMA on tile kb.

#define LDS_STRIDE 40          // 32 + 8 pad elements (80B rows, matches TDM pad)
#define TILE_U16  (128 * LDS_STRIDE)

__device__ __forceinline__ void gemm_tiles(const u16* __restrict__ Ag, const u16* __restrict__ Bg,
                                           int mBase, int nBase, u32 aRows, u32 bRows,
                                           u16* As, u16* Bs, v8f acc[2][4]) {
    const int lane = threadIdx.x & 31;
    const int wave = threadIdx.x >> 5;
    const int wm = wave & 3, wn = wave >> 2;
    const u32 aLds = (u32)(size_t)As;   // low 32 bits of generic shared ptr = LDS offset
    const u32 bLds = (u32)(size_t)Bs;
    const u32 bufBytes = TILE_U16 * 2;

    if (wave == 0)      tdm_load_tile_2d(aLds, Ag + (size_t)mBase * 1024, 1024, aRows);
    else if (wave == 1) tdm_load_tile_2d(bLds, Bg + (size_t)nBase * 1024, 1024, bRows);

    for (int kb = 0; kb < 32; ++kb) {
        const int cur = kb & 1;
        if (kb + 1 < 32) {
            const u32 off = (u32)((kb + 1) & 1) * bufBytes;
            if (wave == 0)
                tdm_load_tile_2d(aLds + off, Ag + (size_t)mBase * 1024 + (kb + 1) * 32, 1024, aRows);
            else if (wave == 1)
                tdm_load_tile_2d(bLds + off, Bg + (size_t)nBase * 1024 + (kb + 1) * 32, 1024, bRows);
            __builtin_amdgcn_s_wait_tensorcnt(1);   // oldest (current tile) complete
        } else {
            __builtin_amdgcn_s_wait_tensorcnt(0);
        }
        __syncthreads();

        const u16* Asb = As + cur * TILE_U16;
        const u16* Bsb = Bs + cur * TILE_U16;
        v16bf a0 = load_fragA(Asb + (wm * 32 + 0)  * LDS_STRIDE, LDS_STRIDE, lane);
        v16bf a1 = load_fragA(Asb + (wm * 32 + 16) * LDS_STRIDE, LDS_STRIDE, lane);
        #pragma unroll
        for (int nt = 0; nt < 4; ++nt) {
            v16bf b = load_fragB(Bsb + (wn * 64 + nt * 16) * LDS_STRIDE, LDS_STRIDE, lane);
            acc[0][nt] = WMMA_BF16(a0, b, acc[0][nt]);
            acc[1][nt] = WMMA_BF16(a1, b, acc[1][nt]);
        }
        __syncthreads();
    }
}

// ---------------- stage 2: QKV GEMM + scatter ----------------
// X[8192,1024] * Wqkv[3072,1024]^T.  Q scaled by 1/sqrt(64)=0.125 here.
// Q,K stored [B,H,T,D]; V stored transposed [B,H,D,T].

__global__ __launch_bounds__(256) void gemm_qkv_kernel(const u16* __restrict__ Xb,
                                                       const u16* __restrict__ Wb,
                                                       u16* __restrict__ Q,
                                                       u16* __restrict__ K,
                                                       u16* __restrict__ Vt) {
    __shared__ __attribute__((aligned(16))) u16 As[2 * TILE_U16];
    __shared__ __attribute__((aligned(16))) u16 Bs[2 * TILE_U16];
    v8f acc[2][4] = {};
    const int mBase = blockIdx.y * 128, nBase = blockIdx.x * 128;
    gemm_tiles(Xb, Wb, mBase, nBase, 8192u, 3072u, As, Bs, acc);

    const int lane = threadIdx.x & 31, wave = threadIdx.x >> 5;
    const int wm = wave & 3, wn = wave >> 2;
    const int half = lane >> 4, ln = lane & 15;
    #pragma unroll
    for (int mt = 0; mt < 2; ++mt)
        #pragma unroll
        for (int nt = 0; nt < 4; ++nt)
            #pragma unroll
            for (int j = 0; j < 8; ++j) {
                int m = mBase + wm * 32 + mt * 16 + j + 8 * half;
                int n = nBase + wn * 64 + nt * 16 + ln;
                float v = acc[mt][nt][j];
                int b = m >> 11, t = m & 2047;
                if (n < 1024) {
                    int hh = n >> 6, d = n & 63;
                    Q[(((size_t)(b * 16 + hh) * 2048) + t) * 64 + d] = f2bf(v * 0.125f);
                } else if (n < 2048) {
                    int nn = n - 1024, hh = nn >> 6, d = nn & 63;
                    K[(((size_t)(b * 16 + hh) * 2048) + t) * 64 + d] = f2bf(v);
                } else {
                    int nn = n - 2048, hh = nn >> 6, d = nn & 63;
                    Vt[(((size_t)(b * 16 + hh) * 64) + d) * 2048 + t] = f2bf(v);
                }
            }
}

// ---------------- stage 3: causal flash attention ----------------
// grid = (T/128, H, B), 8 waves/block, each wave owns 16 query rows.
// Per key-block of 64: S = Q*K^T (WMMA), mask, online softmax,
// P -> per-wave LDS slab (re-layout C->A), O += P*V via WMMA.

#define P_STRIDE 72  // 64 + 8 pad; 144B rows, 16B aligned

__global__ __launch_bounds__(256) void attn_kernel(const u16* __restrict__ Q,
                                                   const u16* __restrict__ K,
                                                   const u16* __restrict__ Vt,
                                                   u16* __restrict__ O) {
    __shared__ __attribute__((aligned(16))) u16 pBuf[8 * 16 * P_STRIDE];
    const int lane = threadIdx.x & 31, wave = threadIdx.x >> 5;
    const int half = lane >> 4, ln = lane & 15;
    const int b = blockIdx.z, h = blockIdx.y;
    const int qBase = blockIdx.x * 128 + wave * 16;

    const u16* Qp  = Q  + (((size_t)(b * 16 + h) * 2048) + qBase) * 64;
    const u16* Kp0 = K  + ((size_t)(b * 16 + h) * 2048) * 64;
    const u16* Vp0 = Vt + ((size_t)(b * 16 + h) * 64) * 2048;
    u16* pW = pBuf + wave * 16 * P_STRIDE;

    v16bf qf0 = load_fragA(Qp + 0,  64, lane);   // d = 0..31 (pre-scaled by 0.125)
    v16bf qf1 = load_fragA(Qp + 32, 64, lane);   // d = 32..63

    float mst[8], lst[8];
    v8f o[4] = {};
    #pragma unroll
    for (int j = 0; j < 8; ++j) { mst[j] = -1e30f; lst[j] = 0.f; }

    const int kbMax = (qBase + 15) >> 6;
    for (int kb = 0; kb <= kbMax; ++kb) {
        if (kb < kbMax)  // warm next key block in GL2
            __builtin_prefetch(Kp0 + (size_t)(kb + 1) * 4096 + lane * 128, 0, 1);
        // ---- S = Q * K^T for 64 keys ----
        v8f S[4] = {};
        const u16* Kp = Kp0 + (size_t)kb * 64 * 64;
        #pragma unroll
        for (int nt = 0; nt < 4; ++nt) {
            v16bf b0 = load_fragB(Kp + nt * 16 * 64 + 0,  64, lane);
            v16bf b1 = load_fragB(Kp + nt * 16 * 64 + 32, 64, lane);
            S[nt] = WMMA_BF16(qf0, b0, S[nt]);
            S[nt] = WMMA_BF16(qf1, b1, S[nt]);
        }
        // ---- causal mask ----
        #pragma unroll
        for (int nt = 0; nt < 4; ++nt) {
            int key = kb * 64 + nt * 16 + ln;
            #pragma unroll
            for (int j = 0; j < 8; ++j) {
                int qr = qBase + j + 8 * half;
                if (key > qr) S[nt][j] = -1e30f;
            }
        }
        // ---- online softmax (row reductions across 16-lane half-wave) ----
        #pragma unroll
        for (int j = 0; j < 8; ++j) {
            float r = fmaxf(fmaxf(S[0][j], S[1][j]), fmaxf(S[2][j], S[3][j]));
            r = fmaxf(r, __shfl_xor(r, 1));
            r = fmaxf(r, __shfl_xor(r, 2));
            r = fmaxf(r, __shfl_xor(r, 4));
            r = fmaxf(r, __shfl_xor(r, 8));
            float mn = fmaxf(mst[j], r);
            float alpha = __expf(mst[j] - mn);
            mst[j] = mn;
            float rs = 0.f;
            #pragma unroll
            for (int nt = 0; nt < 4; ++nt) {
                float e = __expf(S[nt][j] - mn);
                S[nt][j] = e;
                rs += e;
            }
            rs += __shfl_xor(rs, 1);
            rs += __shfl_xor(rs, 2);
            rs += __shfl_xor(rs, 4);
            rs += __shfl_xor(rs, 8);
            lst[j] = lst[j] * alpha + rs;
            #pragma unroll
            for (int dt = 0; dt < 4; ++dt) o[dt][j] *= alpha;
        }
        // ---- P (C layout) -> LDS -> A-fragment layout ----
        #pragma unroll
        for (int nt = 0; nt < 4; ++nt)
            #pragma unroll
            for (int j = 0; j < 8; ++j)
                pW[(j + 8 * half) * P_STRIDE + nt * 16 + ln] = f2bf(S[nt][j]);

        v16bf p0 = load_fragA(pW + 0,  P_STRIDE, lane);  // keys 0..31 of block
        v16bf p1 = load_fragA(pW + 32, P_STRIDE, lane);  // keys 32..63
        // ---- O += P * V (V^T rows contiguous along keys) ----
        const u16* Vp = Vp0 + kb * 64;
        #pragma unroll
        for (int dt = 0; dt < 4; ++dt) {
            v16bf vb0 = load_fragB(Vp + (size_t)dt * 16 * 2048 + 0,  2048, lane);
            v16bf vb1 = load_fragB(Vp + (size_t)dt * 16 * 2048 + 32, 2048, lane);
            o[dt] = WMMA_BF16(p0, vb0, o[dt]);
            o[dt] = WMMA_BF16(p1, vb1, o[dt]);
        }
    }
    // ---- normalize and store O as bf16 [B, T, C] ----
    #pragma unroll
    for (int dt = 0; dt < 4; ++dt)
        #pragma unroll
        for (int j = 0; j < 8; ++j) {
            int t = qBase + j + 8 * half;
            int c = h * 64 + dt * 16 + ln;
            O[((size_t)b * 2048 + t) * 1024 + c] = f2bf(o[dt][j] / lst[j]);
        }
}

// ---------------- stage 4: output GEMM + bias ----------------

__global__ __launch_bounds__(256) void gemm_out_kernel(const u16* __restrict__ Ob,
                                                       const u16* __restrict__ Wb,
                                                       const float* __restrict__ bias,
                                                       float* __restrict__ out) {
    __shared__ __attribute__((aligned(16))) u16 As[2 * TILE_U16];
    __shared__ __attribute__((aligned(16))) u16 Bs[2 * TILE_U16];
    v8f acc[2][4] = {};
    const int mBase = blockIdx.y * 128, nBase = blockIdx.x * 128;
    gemm_tiles(Ob, Wb, mBase, nBase, 8192u, 1024u, As, Bs, acc);

    const int lane = threadIdx.x & 31, wave = threadIdx.x >> 5;
    const int wm = wave & 3, wn = wave >> 2;
    const int half = lane >> 4, ln = lane & 15;
    #pragma unroll
    for (int mt = 0; mt < 2; ++mt)
        #pragma unroll
        for (int nt = 0; nt < 4; ++nt) {
            int n = nBase + wn * 64 + nt * 16 + ln;
            float bv = bias[n];
            #pragma unroll
            for (int j = 0; j < 8; ++j) {
                int m = mBase + wm * 32 + mt * 16 + j + 8 * half;
                out[(size_t)m * 1024 + n] = acc[mt][nt][j] + bv;
            }
        }
}

// ---------------- launch ----------------

extern "C" void kernel_launch(void* const* d_in, const int* in_sizes, int n_in,
                              void* d_out, int out_size, void* d_ws, size_t ws_size,
                              hipStream_t stream) {
    (void)in_sizes; (void)n_in; (void)out_size; (void)ws_size;
    const float* x     = (const float*)d_in[0];   // [4,2048,1024]
    const float* w_qkv = (const float*)d_in[1];   // [3072,1024]
    const float* w_out = (const float*)d_in[2];   // [1024,1024]
    const float* b_out = (const float*)d_in[3];   // [1024]
    float* out = (float*)d_out;                   // [4,2048,1024]

    u16* Xb    = (u16*)d_ws;                            // 8192*1024
    u16* Wqkvb = Xb    + (size_t)8192 * 1024;           // 3072*1024
    u16* Woutb = Wqkvb + (size_t)3072 * 1024;           // 1024*1024
    u16* Qb    = Woutb + (size_t)1024 * 1024;           // 4*16*2048*64
    u16* Kb    = Qb    + (size_t)8388608;
    u16* Vtb   = Kb    + (size_t)8388608;
    u16* Ob    = Vtb   + (size_t)8388608;               // 8192*1024

    cvt_bf16_kernel<<<8192, 256, 0, stream>>>(x,     Xb,    2097152);
    cvt_bf16_kernel<<<3072, 256, 0, stream>>>(w_qkv, Wqkvb, 786432);
    cvt_bf16_kernel<<<1024, 256, 0, stream>>>(w_out, Woutb, 262144);

    gemm_qkv_kernel<<<dim3(24, 64), 256, 0, stream>>>(Xb, Wqkvb, Qb, Kb, Vtb);
    attn_kernel<<<dim3(16, 16, 4), 256, 0, stream>>>(Qb, Kb, Vtb, Ob);
    gemm_out_kernel<<<dim3(8, 64), 256, 0, stream>>>(Ob, Woutb, b_out, out);
}
